// RGAT_22376779612755
// MI455X (gfx1250) — matrix operand order repaired and checked by reading, not compiled, not run on hardware
//
#include <hip/hip_runtime.h>
#include <hip/hip_bf16.h>

// ---------------- problem constants (fixed by reference) ----------------
#define BQ   8      // batch
#define NN   400    // nodes
#define NP   416    // padded node stride for h^T
#define HH   768    // hidden
#define RR   46     // relations
#define LL   2      // layers
#define LN_EPS 1e-5f
#define NEG_INF_F (-1e30f)

// ---------------- WMMA vector types ----------------
typedef __attribute__((ext_vector_type(16))) __bf16 v16bf;
typedef __attribute__((ext_vector_type(8)))  __bf16 v8bf;
typedef __attribute__((ext_vector_type(2)))  __bf16 v2bf;
typedef __attribute__((ext_vector_type(8)))  float  v8f;

// ---------------- tile config ----------------
#define BM 128
#define BN 128
#define BK 32
#define LDA 40   // padded LDS row (elements) for A tile [BM][BK]
#define LDB 40   // padded LDS row (elements) for B^T tile [BN][BK]

__device__ inline v8bf zero8bf() {
  v8bf v;
#pragma unroll
  for (int e = 0; e < 8; ++e) v[e] = (__bf16)0.0f;
  return v;
}

// ---- CDNA5 async global->LDS copy (16B per lane), tracked by ASYNCcnt ----
__device__ inline void async_copy_b128(__bf16* ldsDst, const void* gsrc) {
  unsigned int loff = (unsigned int)(unsigned long long)ldsDst; // low 32b of generic ptr = LDS offset
  asm volatile("global_load_async_to_lds_b128 %0, %1, off"
               :: "v"(loff), "v"(gsrc)
               : "memory");
}
__device__ inline void wait_async0() {
  asm volatile("s_wait_asynccnt 0" ::: "memory");
}

// Generic A tile (BM x BK) row-major via async-to-LDS; zero ds-stores out of range.
__device__ inline void load_tileA_async(const __bf16* __restrict__ gA, int ldga,
                                        int rowBase, int rowDim, int kBase, int kDim,
                                        __bf16* ldsA, int tid) {
#pragma unroll
  for (int p = 0; p < 2; ++p) {
    int idx = (tid + p * 256) * 8;      // 0..4088
    int row = idx >> 5;                 // /32
    int k   = idx & 31;
    int grow = rowBase + row;
    int gk   = kBase + k;
    __bf16* ldst = &ldsA[row * LDA + k];
    if (grow < rowDim && gk < kDim) {
      async_copy_b128(ldst, (const void*)(gA + (size_t)grow * ldga + gk));
    } else {
      *(v8bf*)ldst = zero8bf();
    }
  }
}

// B^T tile (BN cols x BK k) from already-transposed bf16 global (h^T), no guards:
// pure async row-major copy, lands directly in fragB layout ldsBT[n][k].
__device__ inline void load_tileBT_async_full(const __bf16* __restrict__ gBT, int ldgbt,
                                              int colBase, int kBase,
                                              __bf16* ldsBT, int tid) {
#pragma unroll
  for (int p = 0; p < 2; ++p) {
    int idx = (tid + p * 256) * 8;
    int n = idx >> 5;                   // 0..127 (hcol local)
    int k = idx & 31;
    async_copy_b128(&ldsBT[n * LDB + k],
                    (const void*)(gBT + (size_t)(colBase + n) * ldgbt + kBase + k));
  }
}

// B tile (BK x BN) from f32 global (W), convert to bf16, store transposed.
// No bounds checks (K=768, N=768 always full). 8 packed dword LDS stores per thread.
__device__ inline void load_tileB_f32_full(const float* __restrict__ gB, int ldgb,
                                           int kBase, int nBase,
                                           __bf16* ldsBT, int tid) {
  int k0 = (tid >> 4) * 2;              // 0,2,..,30
  int n0 = (tid & 15) * 8;              // 0,8,..,120
  const float* s0 = gB + (size_t)(kBase + k0) * ldgb + nBase + n0;
  const float* s1 = s0 + ldgb;
  v8bf r0, r1;
#pragma unroll
  for (int e = 0; e < 8; ++e) { r0[e] = (__bf16)s0[e]; r1[e] = (__bf16)s1[e]; }
#pragma unroll
  for (int e = 0; e < 8; ++e) {
    v2bf pv = { r0[e], r1[e] };
    *(v2bf*)(&ldsBT[(n0 + e) * LDB + k0]) = pv;
  }
}

// Build A fragment per CDNA5 16-bit A layout:
// lane L, row = L%16, half = L/16; elems 0..7 = K[8*half .. +7], 8..15 = K[16+8*half .. +7]
__device__ inline v16bf fragA(const __bf16* ldsA, int rowLocal, int lane) {
  int half = lane >> 4;
  const __bf16* base = &ldsA[(rowLocal + (lane & 15)) * LDA + 8 * half];
  v8bf lo = *(const v8bf*)base;
  v8bf hi = *(const v8bf*)(base + 16);
  return __builtin_shufflevector(lo, hi, 0,1,2,3,4,5,6,7,8,9,10,11,12,13,14,15);
}

// Build B fragment per CDNA5 16-bit B layout:
// lane L, col = L%16, half = L/16; elems 0..7 = K[8*half .. +7], 8..15 = K[16+8*half .. +7]  (ldsBT is [n][k])
__device__ inline v16bf fragB(const __bf16* ldsBT, int colLocal, int lane) {
  int half = lane >> 4;
  const __bf16* base = &ldsBT[(colLocal + (lane & 15)) * LDB + 8 * half];
  v8bf lo = *(const v8bf*)base;
  v8bf hi = *(const v8bf*)(base + 16);
  return __builtin_shufflevector(lo, hi, 0,1,2,3,4,5,6,7,8,9,10,11,12,13,14,15);
}

// ---------------- kernel 1: cast feature f32 -> bf16 ----------------
__global__ void cast_f32_bf16_kernel(const float* __restrict__ src,
                                     __bf16* __restrict__ dst, int n) {
  int i = blockIdx.x * blockDim.x + threadIdx.x;
  if (i < n) dst[i] = (__bf16)src[i];
}

// ---------------- kernel 2: masked softmax -> att bf16 ----------------
// att[b][r][n][m] = softmax_m( adj[b][n][m][r]*mask[b][m] + (1-mask)*NEG_INF )
__global__ void softmax_kernel(const float* __restrict__ adj,
                               const float* __restrict__ dmask,
                               __bf16* __restrict__ att) {
  int n = blockIdx.x;
  int b = blockIdx.y;
  int wave = threadIdx.x >> 5;
  int lane = threadIdx.x & 31;

  for (int r = wave; r < RR; r += 8) {
    float lv[13];
    float vmax = NEG_INF_F;
#pragma unroll
    for (int i = 0; i < 13; ++i) {
      int m = lane + 32 * i;
      float v = NEG_INF_F;
      if (m < NN) {
        float a = adj[(((size_t)b * NN + n) * NN + m) * RR + r];
        float msk = dmask[b * NN + m];
        v = a * msk + (1.0f - msk) * NEG_INF_F;
      }
      lv[i] = v;
      vmax = fmaxf(vmax, v);
    }
#pragma unroll
    for (int off = 16; off > 0; off >>= 1)
      vmax = fmaxf(vmax, __shfl_xor(vmax, off, 32));

    float sum = 0.0f;
#pragma unroll
    for (int i = 0; i < 13; ++i) {
      int m = lane + 32 * i;
      float e = (m < NN) ? __expf(lv[i] - vmax) : 0.0f;
      lv[i] = e;
      sum += e;
    }
#pragma unroll
    for (int off = 16; off > 0; off >>= 1)
      sum += __shfl_xor(sum, off, 32);
    float inv = 1.0f / sum;

    __bf16* dst = att + (((size_t)b * RR + r) * NN + n) * NN;
#pragma unroll
    for (int i = 0; i < 13; ++i) {
      int m = lane + 32 * i;
      if (m < NN) dst[m] = (__bf16)(lv[i] * inv);
    }
  }
}

// ---------------- kernel 3: hT[b,r] = (x[b] @ W[l,r])^T  (store transposed) ----------------
__global__ void gemm_xw_kernel(const __bf16* __restrict__ X,   // [B][400][768]
                               const float* __restrict__ Wl,   // [R][768][768]
                               __bf16* __restrict__ HoutT) {   // [B][R][768][416]
  __shared__ __bf16 ldsA[2][BM * LDA];
  __shared__ __bf16 ldsB[2][BN * LDB];

  int tileId = blockIdx.x;          // 0..23 (4 M-tiles x 6 N-tiles)
  int r = blockIdx.y;
  int b = blockIdx.z;
  int tm = tileId & 3;
  int tn = tileId >> 2;
  int rowBase = tm * BM;
  int colBase = tn * BN;

  int tid  = threadIdx.x;
  int lane = tid & 31;
  int wave = tid >> 5;
  int waveM = wave & 3;             // 4 waves along M (32 rows each)
  int waveN = wave >> 2;            // 2 waves along N (64 cols each)

  const __bf16* gA = X + (size_t)b * NN * HH;
  const float*  gB = Wl + (size_t)r * HH * HH;

  v8f acc[2][4];
#pragma unroll
  for (int i = 0; i < 2; ++i)
#pragma unroll
    for (int j = 0; j < 4; ++j)
#pragma unroll
      for (int e = 0; e < 8; ++e) acc[i][j][e] = 0.0f;

  const int KSteps = HH / BK;       // 24
  // prologue: tile 0 into buffer 0
  load_tileA_async(gA, HH, rowBase, NN, 0, HH, ldsA[0], tid);
  load_tileB_f32_full(gB, HH, 0, colBase, ldsB[0], tid);
  wait_async0();
  __syncthreads();

  for (int kbi = 0; kbi < KSteps; ++kbi) {
    int cur = kbi & 1;
    int nxt = cur ^ 1;
    if (kbi + 1 < KSteps) {         // issue next tile while computing current
      load_tileA_async(gA, HH, rowBase, NN, (kbi + 1) * BK, HH, ldsA[nxt], tid);
      load_tileB_f32_full(gB, HH, (kbi + 1) * BK, colBase, ldsB[nxt], tid);
    }

    v16bf af[2], bf[4];
#pragma unroll
    for (int i = 0; i < 2; ++i) af[i] = fragA(ldsA[cur], waveM * 32 + i * 16, lane);
#pragma unroll
    for (int j = 0; j < 4; ++j) bf[j] = fragB(ldsB[cur], waveN * 64 + j * 16, lane);
#pragma unroll
    for (int i = 0; i < 2; ++i)
#pragma unroll
      for (int j = 0; j < 4; ++j)
        acc[i][j] = __builtin_amdgcn_wmma_f32_16x16x32_bf16(
            false, af[i], false, bf[j], (short)0, acc[i][j], false, false);

    wait_async0();
    __syncthreads();
  }

  // transposed store: lane's 8 acc elems are contiguous in the node (row) index
  int half = lane >> 4;
  __bf16* out = HoutT + ((size_t)b * RR + r) * ((size_t)HH * NP);
#pragma unroll
  for (int i = 0; i < 2; ++i)
#pragma unroll
    for (int j = 0; j < 4; ++j) {
      int gc = colBase + waveN * 64 + j * 16 + (lane & 15);      // hidden col
      int gr = rowBase + waveM * 32 + i * 16 + 8 * half;         // node row base (mult of 8)
      if (gr < NN) {
        v8bf pk;
#pragma unroll
        for (int e = 0; e < 8; ++e) pk[e] = (__bf16)acc[i][j][e];
        *(v8bf*)(out + (size_t)gc * NP + gr) = pk;
      }
    }
}

// ---------------- kernel 4: addition[b] = sum_r relu( att[b,r] @ h[b,r] ) ----------------
__global__ void gemm_att_kernel(const __bf16* __restrict__ att,   // [B][R][400][400]
                                const __bf16* __restrict__ HbufT, // [B][R][768][416]
                                float* __restrict__ addition) {   // [B][400][768]
  __shared__ __bf16 ldsA[2][BM * LDA];
  __shared__ __bf16 ldsB[2][BN * LDB];

  int tileId = blockIdx.x;
  int b = blockIdx.y;
  int tm = tileId & 3;
  int tn = tileId >> 2;
  int rowBase = tm * BM;
  int colBase = tn * BN;

  int tid  = threadIdx.x;
  int lane = tid & 31;
  int wave = tid >> 5;
  int waveM = wave & 3;
  int waveN = wave >> 2;
  int half = lane >> 4;

  v8f accS[2][4];
#pragma unroll
  for (int i = 0; i < 2; ++i)
#pragma unroll
    for (int j = 0; j < 4; ++j)
#pragma unroll
      for (int e = 0; e < 8; ++e) accS[i][j][e] = 0.0f;

  const int KSteps = (NN + BK - 1) / BK;  // 13

  for (int r = 0; r < RR; ++r) {
    const __bf16* gA  = att   + ((size_t)b * RR + r) * NN * NN;
    const __bf16* gBT = HbufT + ((size_t)b * RR + r) * ((size_t)HH * NP);

    v8f acc[2][4];
#pragma unroll
    for (int i = 0; i < 2; ++i)
#pragma unroll
      for (int j = 0; j < 4; ++j)
#pragma unroll
        for (int e = 0; e < 8; ++e) acc[i][j][e] = 0.0f;

    load_tileA_async(gA, NN, rowBase, NN, 0, NN, ldsA[0], tid);
    load_tileBT_async_full(gBT, NP, colBase, 0, ldsB[0], tid);
    wait_async0();
    __syncthreads();

    for (int kbi = 0; kbi < KSteps; ++kbi) {
      int cur = kbi & 1;
      int nxt = cur ^ 1;
      if (kbi + 1 < KSteps) {
        load_tileA_async(gA, NN, rowBase, NN, (kbi + 1) * BK, NN, ldsA[nxt], tid);
        load_tileBT_async_full(gBT, NP, colBase, (kbi + 1) * BK, ldsB[nxt], tid);
      }

      v16bf af[2], bf[4];
#pragma unroll
      for (int i = 0; i < 2; ++i) af[i] = fragA(ldsA[cur], waveM * 32 + i * 16, lane);
#pragma unroll
      for (int j = 0; j < 4; ++j) bf[j] = fragB(ldsB[cur], waveN * 64 + j * 16, lane);
#pragma unroll
      for (int i = 0; i < 2; ++i)
#pragma unroll
        for (int j = 0; j < 4; ++j)
          acc[i][j] = __builtin_amdgcn_wmma_f32_16x16x32_bf16(
              false, af[i], false, bf[j], (short)0, acc[i][j], false, false);

      wait_async0();
      __syncthreads();
    }

    // per-relation ReLU then accumulate
#pragma unroll
    for (int i = 0; i < 2; ++i)
#pragma unroll
      for (int j = 0; j < 4; ++j)
#pragma unroll
        for (int e = 0; e < 8; ++e)
          accS[i][j][e] += fmaxf(acc[i][j][e], 0.0f);
  }

  float* out = addition + (size_t)b * NN * HH;
#pragma unroll
  for (int i = 0; i < 2; ++i)
#pragma unroll
    for (int j = 0; j < 4; ++j) {
      int gc = colBase + waveN * 64 + j * 16 + (lane & 15);
#pragma unroll
      for (int e = 0; e < 8; ++e) {
        int gr = rowBase + waveM * 32 + i * 16 + e + 8 * half;
        if (gr < NN) out[(size_t)gr * HH + gc] = accS[i][j][e];
      }
    }
}

// ---------------- kernel 5: z = relu(add)/R ; layernorm ; write bf16 x (+ f32 out) ----------------
__global__ void ln_kernel(const float* __restrict__ addition,
                          __bf16* __restrict__ Xb,
                          float* __restrict__ out, int writeOut) {
  __shared__ float s1[256];
  __shared__ float s2[256];
  int row = blockIdx.x;                 // b*400 + n
  int tid = threadIdx.x;
  const float* src = addition + (size_t)row * HH;

  float z[3];
  float sum = 0.0f, sq = 0.0f;
#pragma unroll
  for (int j = 0; j < 3; ++j) {
    int k = tid + 256 * j;
    float v = fmaxf(src[k], 0.0f) * (1.0f / (float)RR);
    z[j] = v;
    sum += v;
    sq  += v * v;
  }
  s1[tid] = sum; s2[tid] = sq;
  __syncthreads();
  for (int st = 128; st > 0; st >>= 1) {
    if (tid < st) { s1[tid] += s1[tid + st]; s2[tid] += s2[tid + st]; }
    __syncthreads();
  }
  float mu  = s1[0] * (1.0f / (float)HH);
  float var = s2[0] * (1.0f / (float)HH) - mu * mu;
  float rs  = rsqrtf(var + LN_EPS);

  __bf16* xdst = Xb + (size_t)row * HH;
  float*  odst = out + (size_t)row * HH;
#pragma unroll
  for (int j = 0; j < 3; ++j) {
    int k = tid + 256 * j;
    float v = (z[j] - mu) * rs;
    xdst[k] = (__bf16)v;
    if (writeOut) odst[k] = v;
  }
}

// ---------------- host launcher ----------------
extern "C" void kernel_launch(void* const* d_in, const int* in_sizes, int n_in,
                              void* d_out, int out_size, void* d_ws, size_t ws_size,
                              hipStream_t stream) {
  const float* adj     = (const float*)d_in[0];
  const float* feature = (const float*)d_in[1];
  const float* dmask   = (const float*)d_in[2];
  const float* W       = (const float*)d_in[3];
  float* out = (float*)d_out;

  // workspace carve-up (all offsets 256B aligned)
  char* ws = (char*)d_ws;
  size_t attBytes = (size_t)BQ * RR * NN * NN * sizeof(__bf16);   // 117,760,000
  size_t hBytes   = (size_t)BQ * RR * HH * NP * sizeof(__bf16);   // 235,143,168 (h^T, padded)
  size_t xBytes   = (size_t)BQ * NN * HH * sizeof(__bf16);        //   4,915,200
  __bf16* att = (__bf16*)ws;                     ws += attBytes;
  __bf16* hT  = (__bf16*)ws;                     ws += hBytes;
  __bf16* xb  = (__bf16*)ws;                     ws += xBytes;
  float*  add = (float*)ws;

  // 1) cast feature -> bf16
  {
    int n = BQ * NN * HH;
    cast_f32_bf16_kernel<<<(n + 255) / 256, 256, 0, stream>>>(feature, xb, n);
  }
  // 2) masked softmax -> att (bf16, [B][R][N][M])
  {
    dim3 grid(NN, BQ);
    softmax_kernel<<<grid, 256, 0, stream>>>(adj, dmask, att);
  }
  // 3) per-layer: gemm1 (-> h^T), gemm2, layernorm
  for (int l = 0; l < LL; ++l) {
    const float* Wl = W + (size_t)l * RR * HH * HH;
    {
      dim3 grid(24, RR, BQ);   // 4 M-tiles * 6 N-tiles
      gemm_xw_kernel<<<grid, 256, 0, stream>>>(xb, Wl, hT);
    }
    {
      dim3 grid(24, BQ);
      gemm_att_kernel<<<grid, 256, 0, stream>>>(att, hT, add);
    }
    {
      ln_kernel<<<BQ * NN, 256, 0, stream>>>(add, xb, out, (l == LL - 1) ? 1 : 0);
    }
  }
}
